// PointGNN_44916767982120
// MI455X (gfx1250) — matrix operand order repaired and checked by reading, not compiled, not run on hardware
//
#include <hip/hip_runtime.h>
#include <hip/hip_bf16.h>

typedef _Float16 half_t;
typedef __attribute__((ext_vector_type(16))) _Float16 v16h;
typedef __attribute__((ext_vector_type(8)))  _Float16 v8h;
typedef __attribute__((ext_vector_type(8)))  float    v8f;

#define MPTS 512
#define NB   2
#define RADIUS 0.05f

__device__ __forceinline__ v16h concat8(v8h lo, v8h hi) {
    return __builtin_shufflevector(lo, hi, 0, 1, 2, 3, 4, 5, 6, 7,
                                   8, 9, 10, 11, 12, 13, 14, 15);
}

// Wave-local LDS ordering point: CDNA5 LDS ops are in-order per wave (DScnt),
// so wave-private RAW through LDS needs only a compiler fence, no s_barrier.
__device__ __forceinline__ void wave_lds_fence() {
    __builtin_amdgcn_fence(__ATOMIC_ACQ_REL, "wavefront");
    __builtin_amdgcn_wave_barrier();
}

// ---------------- WMMA fragment helpers (CDNA5 16x16x32 f16 layouts) ----------------

// A-matrix 16x32 f16 from row-major LDS tile: lane&15 = row M; lane>>4 selects
// the K-halves {0..7,16..23} vs {8..15,24..31}. Two aligned b128 LDS loads.
__device__ __forceinline__ v16h lds_load_A(const half_t* tile, int ldk, int kbase) {
    const int lane = threadIdx.x & 31;
    const int row  = lane & 15;
    const int kof  = (lane >> 4) << 3;  // 0 or 8
    const v8h* p = reinterpret_cast<const v8h*>(tile + row * ldk + kbase + kof);
    return concat8(p[0], p[2]);  // halfs [0..7] and [16..23] from base
}

// B-matrix fragment from pre-swizzled LDS weights: fragment f stores, for each
// lane, its 16 halfs contiguously (lane = ((k>>3)&1)*16 + col, elem = (k>>4)*8 + (k&7)).
__device__ __forceinline__ v16h lds_load_B_swz(const half_t* Wswz, int frag) {
    const int lane = threadIdx.x & 31;
    const v8h* p = reinterpret_cast<const v8h*>(Wswz + (frag << 9) + lane * 16);
    return concat8(p[0], p[1]);
}

// D/C 16x16 f32: lane&15 = col, VGPR v = row v + 8*(lane>>4).
__device__ __forceinline__ void store_tile_f16(half_t* out, int ldn, int n0, v8f acc,
                                               const float* bias, bool relu) {
    const int lane = threadIdx.x & 31;
    const int col  = n0 + (lane & 15);
    const int r0   = (lane >> 4) << 3;
    const float bb = bias[col];
#pragma unroll
    for (int v = 0; v < 8; ++v) {
        float xv = acc[v] + bb;
        if (relu) xv = fmaxf(xv, 0.f);
        out[(r0 + v) * ldn + col] = (half_t)xv;
    }
}

__device__ __forceinline__ void store_tile_f32(float* out, int ldn, int n0, v8f acc,
                                               const float* bias, bool relu) {
    const int lane = threadIdx.x & 31;
    const int col  = n0 + (lane & 15);
    const int r0   = (lane >> 4) << 3;
    const float bb = bias[col];
#pragma unroll
    for (int v = 0; v < 8; ++v) {
        float xv = acc[v] + bb;
        if (relu) xv = fmaxf(xv, 0.f);
        out[(r0 + v) * ldn + col] = xv;
    }
}

// out(16xN) = act( in(16xK) @ W(KxN) + bias ); A fragments hoisted across col tiles.
template <int K, int N>
__device__ __forceinline__ void wmma_layer_f16(const half_t* in, const half_t* Wswz,
                                               const float* bias, half_t* out, bool relu) {
    constexpr int KT = K / 32, CT = N / 16;
    v16h a[KT];
#pragma unroll
    for (int kt = 0; kt < KT; ++kt) a[kt] = lds_load_A(in, K, kt * 32);
#pragma unroll
    for (int ct = 0; ct < CT; ++ct) {
        v8f acc = {0.f, 0.f, 0.f, 0.f, 0.f, 0.f, 0.f, 0.f};
#pragma unroll
        for (int kt = 0; kt < KT; ++kt) {
            v16h b = lds_load_B_swz(Wswz, kt * CT + ct);
            acc = __builtin_amdgcn_wmma_f32_16x16x32_f16(false, a[kt], false, b,
                                                         (short)0, acc, false, false);
        }
        store_tile_f16(out, N, ct * 16, acc, bias, relu);
    }
}

template <int K, int N>
__device__ __forceinline__ void wmma_layer_f32out(const half_t* in, const half_t* Wswz,
                                                  const float* bias, float* out, bool relu) {
    constexpr int KT = K / 32, CT = N / 16;
    v16h a[KT];
#pragma unroll
    for (int kt = 0; kt < KT; ++kt) a[kt] = lds_load_A(in, K, kt * 32);
#pragma unroll
    for (int ct = 0; ct < CT; ++ct) {
        v8f acc = {0.f, 0.f, 0.f, 0.f, 0.f, 0.f, 0.f, 0.f};
#pragma unroll
        for (int kt = 0; kt < KT; ++kt) {
            v16h b = lds_load_B_swz(Wswz, kt * CT + ct);
            acc = __builtin_amdgcn_wmma_f32_16x16x32_f16(false, a[kt], false, b,
                                                         (short)0, acc, false, false);
        }
        store_tile_f32(out, N, ct * 16, acc, bias, relu);
    }
}

// Stage W(K x N) f32 -> LDS f16 in B-fragment-swizzled order (K rows padded from Korig).
template <int Korig, int K, int N>
__device__ __forceinline__ void stage_weights_swz(const float* __restrict__ W,
                                                  half_t* dst, int tid) {
    constexpr int CT = N / 16;
    for (int idx = tid; idx < K * N; idx += 128) {
        const int f = idx >> 9;           // fragment id = kt*CT + ct
        const int r = idx & 511;
        const int lane = r >> 4, q = r & 15;
        const int kt = f / CT, ct = f - kt * CT;
        const int k = kt * 32 + ((q >> 3) << 4) + ((lane >> 4) << 3) + (q & 7);
        const int nn = ct * 16 + (lane & 15);
        const float v = (k < Korig) ? W[k * N + nn] : 0.f;
        dst[idx] = (half_t)v;
    }
}

// ---------------- adjacency init: adj = (||xi-xj||^2 < R) ----------------

__global__ void adj_init_kernel(const float* __restrict__ x, float* __restrict__ adj) {
    int idx = blockIdx.x * blockDim.x + threadIdx.x;
    if (idx >= NB * MPTS * MPTS) return;
    const int j = idx & (MPTS - 1);
    const int i = (idx >> 9) & (MPTS - 1);
    const int n = idx >> 18;
    float d = 0.f;
#pragma unroll
    for (int c = 0; c < 3; ++c) {
        float t = x[(n * MPTS + i) * 3 + c] - x[(n * MPTS + j) * 3 + c];
        d += t * t;
    }
    adj[idx] = (d < RADIUS) ? 1.f : 0.f;
}

// ---------------- tiny per-point MLPs (VALU; negligible FLOPs) ----------------

__global__ __launch_bounds__(128) void point_mlp_h(
    const float* __restrict__ st,
    const float* __restrict__ W1, const float* __restrict__ b1,
    const float* __restrict__ W2, const float* __restrict__ b2,
    const float* __restrict__ W3, const float* __restrict__ b3,
    float* __restrict__ delta) {
    const int p = blockIdx.x, tid = threadIdx.x;
    __shared__ float s_in[3], s1[64], s2[128];
    if (tid < 3) s_in[tid] = st[p * 3 + tid];
    __syncthreads();
    if (tid < 64) {
        float a = b1[tid];
#pragma unroll
        for (int k = 0; k < 3; ++k) a += s_in[k] * W1[k * 64 + tid];
        s1[tid] = fmaxf(a, 0.f);
    }
    __syncthreads();
    {
        float a = b2[tid];
        for (int k = 0; k < 64; ++k) a += s1[k] * W2[k * 128 + tid];
        s2[tid] = fmaxf(a, 0.f);
    }
    __syncthreads();
    if (tid < 3) {
        float a = b3[tid];
        for (int k = 0; k < 128; ++k) a += s2[k] * W3[k * 3 + tid];
        delta[p * 3 + tid] = a;
    }
}

__global__ __launch_bounds__(128) void point_mlp_g(
    const float* __restrict__ agg,
    const float* __restrict__ W1, const float* __restrict__ b1,
    const float* __restrict__ W2, const float* __restrict__ b2,
    const float* __restrict__ W3, const float* __restrict__ b3,
    float* __restrict__ stateCur) {
    const int p = blockIdx.x, tid = threadIdx.x;
    __shared__ float s_in[128], s1[64], s2[32];
    s_in[tid] = agg[p * 128 + tid];
    __syncthreads();
    if (tid < 64) {
        float a = b1[tid];
        for (int k = 0; k < 128; ++k) a += s_in[k] * W1[k * 64 + tid];
        s1[tid] = fmaxf(a, 0.f);
    }
    __syncthreads();
    if (tid < 32) {
        float a = b2[tid];
        for (int k = 0; k < 64; ++k) a += s1[k] * W2[k * 32 + tid];
        s2[tid] = fmaxf(a, 0.f);
    }
    __syncthreads();
    if (tid < 3) {
        float a = b3[tid];
#pragma unroll
        for (int k = 0; k < 32; ++k) a += s2[k] * W3[k * 3 + tid];
        stateCur[p * 3 + tid] += fmaxf(a, 0.f);
    }
}

// ---------------- the big edge kernel ----------------
// Grid: NB*MPTS*2 blocks; block = 4 wave32 owning (n, i, half of j range).
// Each wave loops over 4 j-tiles of 16, running the full f/r pipeline per tile.
// Stages touch only wave-private LDS scratch, so inter-stage sync is a
// wave-local fence (free); the single cross-wave sync is after weight staging.

__global__ __launch_bounds__(128) void edge_kernel(
    const float* __restrict__ x, const float* __restrict__ stateCur,
    const float* __restrict__ delta, float* __restrict__ adj, float* __restrict__ agg,
    const float* __restrict__ fW1, const float* __restrict__ fB1,
    const float* __restrict__ fW2, const float* __restrict__ fB2,
    const float* __restrict__ fW3, const float* __restrict__ fB3,
    const float* __restrict__ rW1, const float* __restrict__ rB1,
    const float* __restrict__ rW2, const float* __restrict__ rB2,
    const float* __restrict__ rW3, const float* __restrict__ rB3) {
    __shared__ alignas(16) half_t s_fW1[32 * 64];     // K padded 6 -> 32, swizzled
    __shared__ alignas(16) half_t s_fW2[64 * 128];
    __shared__ alignas(16) half_t s_fW3[128 * 128];
    __shared__ alignas(16) half_t s_rW1[128 * 64];
    __shared__ alignas(16) half_t s_rW2[64 * 32];
    __shared__ float  s_fB1[64], s_fB2[128], s_fB3[128];
    __shared__ float  s_rB1[64], s_rB2[32], s_rW3v[32];
    __shared__ float  s_rB3;
    __shared__ alignas(16) half_t s_act[4][2][16 * 128];  // per-wave ping-pong tiles
    __shared__ alignas(16) float  s_r2[4][16 * 32];       // r-layer2 f32 per wave
    __shared__ float  s_adjO[4][16], s_adjN[4][16];

    const int tid = threadIdx.x;
    const int bid = blockIdx.x;
    const int jg  = bid & 1;
    const int i   = (bid >> 1) & (MPTS - 1);
    const int n   = bid >> 10;
    const size_t adjRow = (size_t)(n * MPTS + i) * MPTS;

    // Prefetch this block's 1KB adjacency row slice (gfx1250 global_prefetch_b8).
    __builtin_prefetch(adj + adjRow + jg * (MPTS / 2) + tid * 2, 0, 1);

    // Stage all f/r weights once per block into fragment-swizzled f16 LDS.
    stage_weights_swz<6, 32, 64>(fW1, s_fW1, tid);
    stage_weights_swz<64, 64, 128>(fW2, s_fW2, tid);
    stage_weights_swz<128, 128, 128>(fW3, s_fW3, tid);
    stage_weights_swz<128, 128, 64>(rW1, s_rW1, tid);
    stage_weights_swz<64, 64, 32>(rW2, s_rW2, tid);
    if (tid < 64)  s_fB1[tid] = fB1[tid];
    s_fB2[tid] = fB2[tid];
    s_fB3[tid] = fB3[tid];
    if (tid < 64)  s_rB1[tid] = rB1[tid];
    if (tid < 32)  s_rB2[tid] = rB2[tid];
    if (tid < 32)  s_rW3v[tid] = rW3[tid];
    if (tid == 0)  s_rB3 = rB3[0];
    __syncthreads();   // only true cross-wave dependency

    const int w    = tid >> 5;
    const int lane = tid & 31;
    const int base_i = (n * MPTS + i) * 3;
    const float xi0 = x[base_i + 0], xi1 = x[base_i + 1], xi2 = x[base_i + 2];
    const float si0 = stateCur[base_i + 0], si1 = stateCur[base_i + 1],
                si2 = stateCur[base_i + 2];
    float* aggp = agg + (size_t)(n * MPTS + i) * 128;

    for (int it = 0; it < MPTS / 128; ++it) {
        const int j0 = jg * (MPTS / 2) + it * 64 + w * 16;

        // Build masked edge-input tile: 16 rows (j) x 32 cols (6 feats zero-padded).
        // One packed b128 store + 3 zero b128 stores per row (lanes 0-15).
        {
            half_t* tIn = s_act[w][0];
            const v8h zero8 = {};
            if (lane < 16) {
                const int row = lane;
                const int j = j0 + row;
                const float a = adj[adjRow + j];
                const int bj = (n * MPTS + j) * 3;
                v8h pk = {};
                pk[0] = (half_t)((xi0 - x[bj + 0] + delta[bj + 0]) * a);
                pk[1] = (half_t)((xi1 - x[bj + 1] + delta[bj + 1]) * a);
                pk[2] = (half_t)((xi2 - x[bj + 2] + delta[bj + 2]) * a);
                pk[3] = (half_t)(si0 * a);
                pk[4] = (half_t)(si1 * a);
                pk[5] = (half_t)(si2 * a);
                v8h* rp = reinterpret_cast<v8h*>(tIn + row * 32);
                rp[0] = pk;
                rp[1] = zero8;
                rp[2] = zero8;
                rp[3] = zero8;
                s_adjO[w][row] = a;
            }
        }
        wave_lds_fence();

        // f MLP: [6(pad32) -> 64 -> 128 -> 128], relu each
        wmma_layer_f16<32, 64>(s_act[w][0], s_fW1, s_fB1, s_act[w][1], true);
        wave_lds_fence();
        wmma_layer_f16<64, 128>(s_act[w][1], s_fW2, s_fB2, s_act[w][0], true);
        wave_lds_fence();
        wmma_layer_f16<128, 128>(s_act[w][0], s_fW3, s_fB3, s_act[w][1], true);  // eij
        wave_lds_fence();
        // r MLP layers 1-2: [128 -> 64 -> 32], relu, relu
        wmma_layer_f16<128, 64>(s_act[w][1], s_rW1, s_rB1, s_act[w][0], true);
        wave_lds_fence();
        wmma_layer_f32out<64, 32>(s_act[w][0], s_rW2, s_rB2, s_r2[w], true);
        wave_lds_fence();

        // r layer 3 (32 -> 1) + hardsigmoid adjacency update; lanes 0-15.
        if (lane < 16) {
            float acc = s_rB3;
#pragma unroll
            for (int k = 0; k < 32; ++k) acc += s_r2[w][lane * 32 + k] * s_rW3v[k];
            const float an =
                fminf(fmaxf((acc + s_adjO[w][lane]) * (1.f / 6.f) + 0.5f, 0.f), 1.f);
            s_adjN[w][lane] = an;
            adj[adjRow + j0 + lane] = an;  // (n,i,j) owned solely by this wave
        }
        wave_lds_fence();

        // agg[n,i,:] = max_j eij * adjNew. Lane owns 8 contiguous features
        // (c = lane&15) over 8 rows (lane>>4 half); combine halves via shfl_xor.
        {
            const half_t* eij = s_act[w][1];
            const int c = lane & 15;
            const int rb = (lane >> 4) << 3;
            float m[8] = {0.f, 0.f, 0.f, 0.f, 0.f, 0.f, 0.f, 0.f};
#pragma unroll
            for (int r = 0; r < 8; ++r) {
                const v8h e = *reinterpret_cast<const v8h*>(eij + (rb + r) * 128 + c * 8);
                const float aN = s_adjN[w][rb + r];
#pragma unroll
                for (int t = 0; t < 8; ++t) m[t] = fmaxf(m[t], (float)e[t] * aN);
            }
#pragma unroll
            for (int t = 0; t < 8; ++t) m[t] = fmaxf(m[t], __shfl_xor(m[t], 16, 32));
            if (lane < 16) {
#pragma unroll
                for (int t = 0; t < 8; ++t)
                    atomicMax((unsigned int*)(aggp + c * 8 + t), __float_as_uint(m[t]));
            }
        }
        wave_lds_fence();
    }
}

// ---------------- host orchestration ----------------
// d_in: [0]=x, [1]=state, then per t (dict keys sorted f,g,h,r), each layer (W,b):
//   f: +0..5   g: +6..11   h: +12..17   r: +18..23

extern "C" void kernel_launch(void* const* d_in, const int* in_sizes, int n_in,
                              void* d_out, int out_size, void* d_ws, size_t ws_size,
                              hipStream_t stream) {
    (void)in_sizes; (void)n_in; (void)out_size; (void)ws_size;
    const float* x  = (const float*)d_in[0];
    const float* st = (const float*)d_in[1];
    auto P = [&](int t, int k) -> const float* {
        return (const float*)d_in[2 + t * 24 + k];
    };

    float* ws       = (float*)d_ws;
    float* adj      = ws;                       // NB*MPTS*MPTS = 524288 f
    float* delta    = adj + NB * MPTS * MPTS;   // 3072 f
    float* agg      = delta + NB * MPTS * 3;    // NB*MPTS*128 = 131072 f
    float* stateCur = agg + NB * MPTS * 128;    // 3072 f

    hipMemcpyAsync(stateCur, st, NB * MPTS * 3 * sizeof(float),
                   hipMemcpyDeviceToDevice, stream);
    adj_init_kernel<<<(NB * MPTS * MPTS + 255) / 256, 256, 0, stream>>>(x, adj);

    for (int t = 0; t < 3; ++t) {
        point_mlp_h<<<NB * MPTS, 128, 0, stream>>>(
            stateCur, P(t, 12), P(t, 13), P(t, 14), P(t, 15), P(t, 16), P(t, 17), delta);
        hipMemsetAsync(agg, 0, NB * MPTS * 128 * sizeof(float), stream);
        edge_kernel<<<NB * MPTS * 2, 128, 0, stream>>>(
            x, stateCur, delta, adj, agg,
            P(t, 0), P(t, 1), P(t, 2), P(t, 3), P(t, 4), P(t, 5),
            P(t, 18), P(t, 19), P(t, 20), P(t, 21), P(t, 22), P(t, 23));
        point_mlp_g<<<NB * MPTS, 128, 0, stream>>>(
            agg, P(t, 6), P(t, 7), P(t, 8), P(t, 9), P(t, 10), P(t, 11), stateCur);
    }
    hipMemcpyAsync(d_out, stateCur, NB * MPTS * 3 * sizeof(float),
                   hipMemcpyDeviceToDevice, stream);
}
